// PoHBlock_481036337477
// MI455X (gfx1250) — matrix operand order repaired and checked by reading, not compiled, not run on hardware
//
#include <hip/hip_runtime.h>

typedef unsigned short u16;
typedef unsigned int   u32;
typedef __attribute__((ext_vector_type(16))) __bf16 bf16x16;
typedef __attribute__((ext_vector_type(8)))  float  f32x8;

#define DMODEL 1024
#define NHEAD  16
#define DHEAD  64
#define DFF    4096
#define BATCH  8
#define SEQ    1024
#define BT     (BATCH*SEQ)   // 8192

__device__ __forceinline__ u16 f2bf(float f) {
  u32 u = __builtin_bit_cast(u32, f);
  u32 r = u + 0x7FFFu + ((u >> 16) & 1u);   // round-to-nearest-even
  return (u16)(r >> 16);
}

union FragAB { uint4 q[2]; bf16x16 v; };
union FragC  { f32x8 v; float f[8]; };

#define WMMA_BF16(A, B, C) \
  __builtin_amdgcn_wmma_f32_16x16x32_bf16(false, (A), false, (B), (short)0, (C), false, false)

// ---------------------------------------------------------------------------
// Generic GEMM: C[M,N] = A[M,K](bf16 row-major) * Bt[N,K](bf16, B transposed)
//               + bias[N], optional relu, fp32 or bf16 output.
// Block: 256 thr = 8 waves; block tile 128x64; each wave 32x32 (2x2 wmma tiles)
// ---------------------------------------------------------------------------
template<int OUT_BF16, int RELU>
__global__ __launch_bounds__(256) void gemm_bf16_wmma(
    const u16* __restrict__ A, const u16* __restrict__ Bt,
    const float* __restrict__ bias, void* __restrict__ Cout,
    int M, int N, int K)
{
  const int lane = threadIdx.x & 31;
  const int wave = threadIdx.x >> 5;
  const int h = lane >> 4, r = lane & 15;
  const int wm = wave & 3, wn = wave >> 2;
  const int m0 = blockIdx.y * 128 + wm * 32;
  const int n0 = blockIdx.x * 64  + wn * 32;

  FragC acc[2][2];
  for (int a = 0; a < 2; a++)
    for (int bq2 = 0; bq2 < 2; bq2++)
      for (int i = 0; i < 8; i++) acc[a][bq2].f[i] = 0.f;

  const u16* arow0 = A  + (size_t)(m0 + r)      * K;
  const u16* arow1 = A  + (size_t)(m0 + 16 + r) * K;
  const u16* brow0 = Bt + (size_t)(n0 + r)      * K;
  const u16* brow1 = Bt + (size_t)(n0 + 16 + r) * K;

  for (int k0 = 0; k0 < K; k0 += 32) {
    FragAB a0, a1, b0, b1;
    // A fragment: lane half h -> K chunks {k0+8h..+7, k0+16+8h..+7}
    a0.q[0] = *(const uint4*)(arow0 + k0 + 8*h);
    a0.q[1] = *(const uint4*)(arow0 + k0 + 16 + 8*h);
    a1.q[0] = *(const uint4*)(arow1 + k0 + 8*h);
    a1.q[1] = *(const uint4*)(arow1 + k0 + 16 + 8*h);
    // B fragment: lane half h -> K chunks {k0+8h..+7, k0+16+8h..+7} of column n
    b0.q[0] = *(const uint4*)(brow0 + k0 + 8*h);
    b0.q[1] = *(const uint4*)(brow0 + k0 + 16 + 8*h);
    b1.q[0] = *(const uint4*)(brow1 + k0 + 8*h);
    b1.q[1] = *(const uint4*)(brow1 + k0 + 16 + 8*h);

    acc[0][0].v = WMMA_BF16(a0.v, b0.v, acc[0][0].v);
    acc[0][1].v = WMMA_BF16(a0.v, b1.v, acc[0][1].v);
    acc[1][0].v = WMMA_BF16(a1.v, b0.v, acc[1][0].v);
    acc[1][1].v = WMMA_BF16(a1.v, b1.v, acc[1][1].v);
  }

  for (int im = 0; im < 2; im++)
    for (int in_ = 0; in_ < 2; in_++)
      for (int i = 0; i < 8; i++) {
        int row = m0 + im * 16 + i + 8 * h;     // C layout: vgpr i -> row i+8h
        int col = n0 + in_ * 16 + r;            //          lane&15 -> col
        float vv = acc[im][in_].f[i] + bias[col];
        if (RELU) vv = vv > 0.f ? vv : 0.f;
        if (OUT_BF16) ((u16*)Cout)[(size_t)row * N + col] = f2bf(vv);
        else          ((float*)Cout)[(size_t)row * N + col] = vv;
      }
}

// ---------------------------------------------------------------------------
// Fused flash attention. qkv: [B*T, 3072] bf16 (q | k | v, head-major cols).
// out: [B*T, 1024] bf16. Block = 128 thr = 4 waves; each wave owns 16 q-rows.
// K tile is staged with gfx1250 async global->LDS copies (ASYNCcnt).
// ---------------------------------------------------------------------------
__global__ __launch_bounds__(128) void flash_attn_wmma(
    const u16* __restrict__ qkv, u16* __restrict__ out)
{
  __shared__ alignas(16) u16 Kt[32 * 64];       // [key][dh] row-major
  __shared__ alignas(16) u16 Vt[64 * 48];       // [dh][key], padded stride 48
  __shared__ alignas(16) u16 Ps[4][16 * 32];    // per-wave P staging

  const int tid  = threadIdx.x;
  const int lane = tid & 31, wave = tid >> 5;
  const int h = lane >> 4, r = lane & 15;
  const int bh = blockIdx.y;
  const int b = bh >> 4, hh = bh & 15;
  const int q0 = blockIdx.x * 64 + wave * 16;
  const float scale = 0.125f;                   // 1/sqrt(64)

  // Q fragments for this wave's 16 rows, DH=64 -> 2 K-chunks of 32
  FragAB Qa[2];
  {
    const u16* qp = qkv + (size_t)(b * SEQ + q0 + r) * 3072 + hh * 64;
    for (int c = 0; c < 2; c++) {
      Qa[c].q[0] = *(const uint4*)(qp + 32 * c + 8 * h);
      Qa[c].q[1] = *(const uint4*)(qp + 32 * c + 16 + 8 * h);
    }
  }

  // uniform SGPR base for the K rows of this (b, head)
  const u16* kbase = qkv + (size_t)b * SEQ * 3072 + 1024 + hh * 64;

  float mstate[8], lstate[8];
  FragC Oa[4];
  for (int i = 0; i < 8; i++) { mstate[i] = -1e30f; lstate[i] = 0.f; }
  for (int t = 0; t < 4; t++) for (int i = 0; i < 8; i++) Oa[t].f[i] = 0.f;

  for (int kt = 0; kt < SEQ; kt += 32) {
    __syncthreads();
    // --- K tile: async global -> LDS (no VGPR staging), 16B per copy -------
    for (int idx = tid; idx < 256; idx += 128) {
      int row = idx >> 3, cc = idx & 7;
      u32 goff = ((u32)(kt + row) * 3072u + (u32)cc * 8u) * 2u;   // bytes
      u32 loff = (u32)(uintptr_t)(&Kt[row * 64 + cc * 8]);        // LDS bytes
      asm volatile("global_load_async_to_lds_b128 %0, %1, %2 offset:0"
                   :: "v"(loff), "v"(goff), "s"(kbase) : "memory");
    }
    // --- V tile: VMEM load + transposed LDS store --------------------------
    for (int idx = tid; idx < 256; idx += 128) {
      int row = idx >> 3, cc = idx & 7;
      const u16* vp = qkv + (size_t)(b * SEQ + kt + row) * 3072 + 2048 + hh * 64 + cc * 8;
      uint4 vd = *(const uint4*)vp;
      const u16* ve = (const u16*)&vd;
      for (int j = 0; j < 8; j++) Vt[(cc * 8 + j) * 48 + row] = ve[j];
    }
    asm volatile("s_wait_asynccnt 0" ::: "memory");
    __syncthreads();

    // S = Q * K^T  (16 x 32 keys), two 16x16 C tiles
    FragC S0, S1;
    for (int i = 0; i < 8; i++) { S0.f[i] = 0.f; S1.f[i] = 0.f; }
    for (int c = 0; c < 2; c++) {
      FragAB bk0, bk1;    // B = K^T: column n = key, K-dim = dh (contig in Kt row)
      bk0.q[0] = *(const uint4*)(&Kt[r * 64 + 32 * c + 8 * h]);
      bk0.q[1] = *(const uint4*)(&Kt[r * 64 + 32 * c + 16 + 8 * h]);
      bk1.q[0] = *(const uint4*)(&Kt[(16 + r) * 64 + 32 * c + 8 * h]);
      bk1.q[1] = *(const uint4*)(&Kt[(16 + r) * 64 + 32 * c + 16 + 8 * h]);
      S0.v = WMMA_BF16(Qa[c].v, bk0.v, S0.v);
      S1.v = WMMA_BF16(Qa[c].v, bk1.v, S1.v);
    }

    // online softmax: row = i + 8h, row values live in lanes of the same half
    for (int i = 0; i < 8; i++) {
      float s0 = S0.f[i] * scale, s1 = S1.f[i] * scale;
      float m = fmaxf(s0, s1);
      for (int mask = 1; mask < 16; mask <<= 1) m = fmaxf(m, __shfl_xor(m, mask, 32));
      float mnew  = fmaxf(mstate[i], m);
      float alpha = __expf(mstate[i] - mnew);
      mstate[i] = mnew;
      float p0 = __expf(s0 - mnew), p1 = __expf(s1 - mnew);
      float rs = p0 + p1;
      for (int mask = 1; mask < 16; mask <<= 1) rs += __shfl_xor(rs, mask, 32);
      lstate[i] = lstate[i] * alpha + rs;
      for (int t = 0; t < 4; t++) Oa[t].f[i] *= alpha;
      Ps[wave][(i + 8 * h) * 32 + r]      = f2bf(p0);
      Ps[wave][(i + 8 * h) * 32 + 16 + r] = f2bf(p1);
    }

    // reload P in A-fragment layout (same-wave LDS ops are ordered)
    FragAB Pa;
    Pa.q[0] = *(const uint4*)(&Ps[wave][r * 32 + 8 * h]);
    Pa.q[1] = *(const uint4*)(&Ps[wave][r * 32 + 16 + 8 * h]);

    // O += P * V   (B = V: column n = dh, K-dim = key, contig in Vt row)
    for (int t = 0; t < 4; t++) {
      FragAB bv;
      bv.q[0] = *(const uint4*)(&Vt[(t * 16 + r) * 48 + 8 * h]);
      bv.q[1] = *(const uint4*)(&Vt[(t * 16 + r) * 48 + 16 + 8 * h]);
      Oa[t].v = WMMA_BF16(Pa.v, bv.v, Oa[t].v);
    }
  }

  for (int t = 0; t < 4; t++)
    for (int i = 0; i < 8; i++) {
      int trow = q0 + i + 8 * h;
      out[(size_t)(b * SEQ + trow) * 1024 + hh * 64 + t * 16 + r] =
          f2bf(Oa[t].f[i] / lstate[i]);
    }
}

// ---------------------------------------------------------------------------
// z = LayerNorm(z + y) * g + be ; also writes bf16 shadow. One block per row.
// ---------------------------------------------------------------------------
__global__ __launch_bounds__(256) void add_ln_kernel(
    const float* x, const float* __restrict__ yadd,
    const float* __restrict__ g, const float* __restrict__ be,
    float* zout, u16* __restrict__ zbf)
{
  __shared__ float red[256];
  const int tid = threadIdx.x;
  const size_t base = (size_t)blockIdx.x * 1024;
  float v[4]; float s = 0.f;
  for (int j = 0; j < 4; j++) {
    int c = tid + j * 256;
    v[j] = x[base + c] + yadd[base + c];
    s += v[j];
  }
  red[tid] = s; __syncthreads();
  for (int off = 128; off > 0; off >>= 1) { if (tid < off) red[tid] += red[tid + off]; __syncthreads(); }
  float mean = red[0] * (1.0f / 1024.0f);
  __syncthreads();
  float sq = 0.f;
  for (int j = 0; j < 4; j++) { float d = v[j] - mean; sq += d * d; }
  red[tid] = sq; __syncthreads();
  for (int off = 128; off > 0; off >>= 1) { if (tid < off) red[tid] += red[tid + off]; __syncthreads(); }
  float rstd = rsqrtf(red[0] * (1.0f / 1024.0f) + 1e-5f);
  for (int j = 0; j < 4; j++) {
    int c = tid + j * 256;
    float o = g[c] * (v[j] - mean) * rstd + be[c];
    zout[base + c] = o;
    zbf[base + c]  = f2bf(o);
  }
}

// ------------------------- prep kernels ------------------------------------
__global__ void init_z_kernel(const float* __restrict__ zin,
                              float* __restrict__ z, u16* __restrict__ zbf, int n) {
  int i = blockIdx.x * 256 + threadIdx.x;
  if (i < n) { float v = zin[i]; z[i] = v; zbf[i] = f2bf(v); }
}

// Wq/Wk/Wv (H,D,DH) -> Wqkv_t[3072][1024] bf16: row n = mat*1024 + h*64 + e
__global__ void prep_wqkv_kernel(const float* __restrict__ Wq,
                                 const float* __restrict__ Wk,
                                 const float* __restrict__ Wv,
                                 u16* __restrict__ dst) {
  int i = blockIdx.x * 256 + threadIdx.x;          // 3*2^20
  int mat  = i >> 20;
  int remn = i & 0xFFFFF;
  int n = remn >> 10;
  int d = remn & 1023;
  int hh = n >> 6, e = n & 63;
  const float* W = (mat == 0) ? Wq : ((mat == 1) ? Wk : Wv);
  dst[i] = f2bf(W[(size_t)hh * 65536 + d * 64 + e]);
}

__global__ void prep_bias_kernel(const float* __restrict__ bq,
                                 const float* __restrict__ bk,
                                 const float* __restrict__ bv,
                                 float* __restrict__ dst) {
  int i = blockIdx.x * 256 + threadIdx.x;          // 3072
  dst[i] = (i < 1024) ? bq[i] : ((i < 2048) ? bk[i - 1024] : bv[i - 2048]);
}

// src[K][N] fp32 -> dst[N][K] bf16
__global__ void transpose_bf16_kernel(const float* __restrict__ src,
                                      u16* __restrict__ dst, int K, int N) {
  int i = blockIdx.x * 256 + threadIdx.x;          // N*K
  int n = i / K, k = i - n * K;
  dst[i] = f2bf(src[(size_t)k * N + n]);
}

// ---------------------------------------------------------------------------
extern "C" void kernel_launch(void* const* d_in, const int* in_sizes, int n_in,
                              void* d_out, int out_size, void* d_ws, size_t ws_size,
                              hipStream_t stream)
{
  const float* z_in = (const float*)d_in[0];
  const float* Wq   = (const float*)d_in[1];
  const float* bq   = (const float*)d_in[2];
  const float* Wk   = (const float*)d_in[3];
  const float* bk   = (const float*)d_in[4];
  const float* Wv   = (const float*)d_in[5];
  const float* bv   = (const float*)d_in[6];
  const float* Wo   = (const float*)d_in[7];
  const float* bo   = (const float*)d_in[8];
  const float* W1   = (const float*)d_in[9];
  const float* b1   = (const float*)d_in[10];
  const float* W2   = (const float*)d_in[11];
  const float* b2   = (const float*)d_in[12];
  const float* g1   = (const float*)d_in[13];
  const float* be1  = (const float*)d_in[14];
  const float* g2   = (const float*)d_in[15];
  const float* be2  = (const float*)d_in[16];

  char* w = (char*)d_ws;
  size_t off = 0;
  auto alloc = [&](size_t bytes) {
    void* p = w + off; off += (bytes + 255) & ~((size_t)255); return p;
  };
  float* z      = (float*)alloc((size_t)BT * 1024 * 4);   // 32 MB
  u16*   z_bf   = (u16*)  alloc((size_t)BT * 1024 * 2);   // 16 MB
  u16*   qkv    = (u16*)  alloc((size_t)BT * 3072 * 2);   // 48 MB
  u16*   attn   = (u16*)  alloc((size_t)BT * 1024 * 2);   // 16 MB
  float* tmp    = (float*)alloc((size_t)BT * 1024 * 4);   // 32 MB
  u16*   hbuf   = (u16*)  alloc((size_t)BT * 4096 * 2);   // 64 MB
  u16*   Wqkv_t = (u16*)  alloc((size_t)3072 * 1024 * 2);
  u16*   Wo_t   = (u16*)  alloc((size_t)1024 * 1024 * 2);
  u16*   W1_t   = (u16*)  alloc((size_t)4096 * 1024 * 2);
  u16*   W2_t   = (u16*)  alloc((size_t)1024 * 4096 * 2);
  float* bcat   = (float*)alloc(3072 * 4);

  init_z_kernel<<<(BT * 1024) / 256, 256, 0, stream>>>(z_in, z, z_bf, BT * 1024);
  prep_wqkv_kernel<<<(3 * 1024 * 1024) / 256, 256, 0, stream>>>(Wq, Wk, Wv, Wqkv_t);
  prep_bias_kernel<<<3072 / 256, 256, 0, stream>>>(bq, bk, bv, bcat);
  transpose_bf16_kernel<<<(1024 * 1024) / 256, 256, 0, stream>>>(Wo, Wo_t, 1024, 1024);
  transpose_bf16_kernel<<<(4096 * 1024) / 256, 256, 0, stream>>>(W1, W1_t, 1024, 4096);
  transpose_bf16_kernel<<<(1024 * 4096) / 256, 256, 0, stream>>>(W2, W2_t, 4096, 1024);

  for (int it = 0; it < 3; it++) {
    // fused QKV projection: [8192,1024] x [1024,3072]
    gemm_bf16_wmma<1, 0><<<dim3(3072 / 64, BT / 128), 256, 0, stream>>>(
        z_bf, Wqkv_t, bcat, qkv, BT, 3072, 1024);
    // attention, all heads
    flash_attn_wmma<<<dim3(SEQ / 64, BATCH * NHEAD), 128, 0, stream>>>(qkv, attn);
    // output projection
    gemm_bf16_wmma<0, 0><<<dim3(1024 / 64, BT / 128), 256, 0, stream>>>(
        attn, Wo_t, bo, tmp, BT, 1024, 1024);
    // z = LN(z + attn_out)  (this is z_refined on the last iteration)
    add_ln_kernel<<<BT, 256, 0, stream>>>(z, tmp, g1, be1, z, z_bf);
    if (it < 2) {
      gemm_bf16_wmma<1, 1><<<dim3(4096 / 64, BT / 128), 256, 0, stream>>>(
          z_bf, W1_t, b1, hbuf, BT, 4096, 1024);
      gemm_bf16_wmma<0, 0><<<dim3(1024 / 64, BT / 128), 256, 0, stream>>>(
          hbuf, W2_t, b2, tmp, BT, 1024, 4096);
      add_ln_kernel<<<BT, 256, 0, stream>>>(z, tmp, g2, be2, z, z_bf);
    }
  }
  hipMemcpyAsync(d_out, z, (size_t)BT * 1024 * 4, hipMemcpyDeviceToDevice, stream);
}
